// MambaBlock_59528246723223
// MI455X (gfx1250) — compile-verified, matrix operand order checked by reading, not run on hardware
//
#include <hip/hip_runtime.h>
#include <hip/hip_bf16.h>
#include <math.h>

// ---------------------------------------------------------------------------
// Mamba block for MI455X (gfx1250, wave32).
// GEMMs: v_wmma_f32_16x16x32_bf16; B tiles staged block-wide into LDS with
// double-buffered global_load_async_to_lds_b128 (ASYNCcnt); A fragments
// register-ping-ponged one K-step ahead (manual 2x unroll: no rotation movs,
// no WMMA->VALU hazard nops). Scan: state-parallel with shuffle reduction.
// ---------------------------------------------------------------------------

typedef __bf16 bf16_t;
typedef bf16_t v16bf __attribute__((ext_vector_type(16)));
typedef float  v8f   __attribute__((ext_vector_type(8)));

#define DIM      1024
#define D_STATE  16
#define D_CONV   4
#define D_INNER  2048
#define DT_RANK  64
#define BATCH    2
#define SEQ      1024
#define NTOK     (BATCH * SEQ)            // 2048 rows in every GEMM
#define XPROJ_N  (DT_RANK + 2 * D_STATE)  // 96

#define BN       64                       // N-tile per block
#define BK       32                       // K-chunk per step
#define PAD_ROW  40                       // bf16 elems per LDS row (32 + 8 pad)

// ---- gfx1250 async global->LDS copy (ASYNCcnt-tracked), GVS addressing ----
static __device__ __forceinline__ void async_load_b128(unsigned lds_addr,
                                                       unsigned gl_off,
                                                       const void* base) {
    asm volatile("global_load_async_to_lds_b128 %0, %1, %2 offset:0"
                 :: "v"(lds_addr), "v"(gl_off), "s"(base)
                 : "memory");
}
static __device__ __forceinline__ void wait_async_le1() {
    asm volatile("s_wait_asynccnt 0x1" ::: "memory");
}

// -------------------- f32 -> bf16 bulk convert (8 elems/thread) ------------
__global__ __launch_bounds__(256)
void cvt8_f32_bf16(const float* __restrict__ src, bf16_t* __restrict__ dst, int n) {
    const int i = (blockIdx.x * 256 + threadIdx.x) * 8;
    if (i >= n) return;
    const float4 a = ((const float4*)(src + i))[0];
    const float4 b = ((const float4*)(src + i))[1];
    union { bf16_t h[8]; uint4 u; } pk;
    pk.h[0] = (bf16_t)a.x; pk.h[1] = (bf16_t)a.y; pk.h[2] = (bf16_t)a.z; pk.h[3] = (bf16_t)a.w;
    pk.h[4] = (bf16_t)b.x; pk.h[5] = (bf16_t)b.y; pk.h[6] = (bf16_t)b.z; pk.h[7] = (bf16_t)b.w;
    *(uint4*)(dst + i) = pk.u;
}

// Per-lane WMMA fragment: elements K {0..7} at p, {16..23} at p+16
// (ISA 7.12.2 16-bit A/B layout). Two b128 loads (global or LDS).
static __device__ __forceinline__ v16bf load_frag_bf16(const bf16_t* __restrict__ p) {
    union { uint4 u[2]; v16bf v; } f;
    f.u[0] = *(const uint4*)p;
    f.u[1] = *(const uint4*)(p + 16);
    return f.v;
}

// One K-step: prefetch next B chunk (async->LDS) + next A fragment, wait for
// the current buffer, issue all 8 ds_load_b128, then 4 back-to-back WMMAs.
static __device__ __forceinline__ void gemm_step(
    int j, int nk, const bf16_t* __restrict__ sb_cur, unsigned lds_next,
    unsigned gbase, const void* __restrict__ Wb, const bf16_t* __restrict__ ap,
    int l16, int koff, const v16bf& af_cur, v16bf& af_next, v8f acc[4]) {
    // Wrap on the final step to keep the ASYNCcnt discipline uniform; the
    // extra tile/fragment is never consumed.
    const int knext = (j + 1 < nk) ? (j + 1) * BK : 0;
    async_load_b128(lds_next, gbase + (unsigned)(knext * 2), Wb);
    af_next = load_frag_bf16(ap + knext);   // latency hides under this step

    wait_async_le1();              // my slice of the current buffer has landed
    __syncthreads();               // ... and everyone else's

    v16bf bfv[4];
#pragma unroll
    for (int t = 0; t < 4; ++t)
        bfv[t] = load_frag_bf16(sb_cur + (t * 16 + l16) * PAD_ROW + koff);
#pragma unroll
    for (int t = 0; t < 4; ++t)
        acc[t] = __builtin_amdgcn_wmma_f32_16x16x32_bf16(
            false, af_cur, false, bfv[t], (short)0, acc[t], false, false);

    __syncthreads();               // all reads done before buffer is rewritten
}

// C[M,N] = A[M,K] (bf16 row-major, lda) x W[N,K]^T (bf16 row-major, ldb).
// Block: 256 thr = 8 waves, 128(M) x 64(N). Per K-step the block async-stages
// the shared 64x32 B tile into LDS once (1 x b128 slice per thread); waves
// read fragments via ds_load_b128 from padded rows (80B stride, conflict-free).
// K-loop manually unrolled 2x with ping-ponged A fragments / LDS buffers.
// mode 0: f32 store. mode 1: softplus(v+bias[col]). mode 2: f32 + bf16 dual.
// Note: all K used here give even nk (32, 64, 2).
__global__ __launch_bounds__(256)
void gemm_bf16_wmma(const bf16_t* __restrict__ A, const bf16_t* __restrict__ W,
                    float* __restrict__ C, bf16_t* __restrict__ Cbf,
                    const float* __restrict__ bias,
                    int M, int N, int K, int lda, int ldb, int ldc, int mode) {
    __shared__ bf16_t sB[2][BN * PAD_ROW];   // 2 x 5120 B

    const int tid     = threadIdx.x;
    const int lane    = tid & 31;
    const int wave    = tid >> 5;
    const int l16     = lane & 15;
    const int halfsel = lane >> 4;           // 0: lanes 0-15, 1: lanes 16-31
    const int koff    = halfsel << 3;        // K sub-offset 0 or 8
    const int mtile   = blockIdx.x * 128 + wave * 16;
    const int nbase   = blockIdx.y * BN;
    const int arow    = mtile + l16;

    // B-staging mapping: thread -> (row 0..63, 16B quarter 0..3) of the tile.
    const int srow = tid >> 2;
    const int sq   = tid & 3;
    const int wrow = (nbase + srow < N) ? (nbase + srow) : (N - 1); // clamp overhang
    const unsigned gbase = (unsigned)(((size_t)wrow * ldb + sq * 8) * 2); // bytes
    const unsigned lds0  = (unsigned)(size_t)&sB[0][srow * PAD_ROW + sq * 8];
    const unsigned lds1  = (unsigned)(size_t)&sB[1][srow * PAD_ROW + sq * 8];

    v8f acc[4];
#pragma unroll
    for (int t = 0; t < 4; ++t)
#pragma unroll
        for (int r = 0; r < 8; ++r) acc[t][r] = 0.0f;

    const int nk = K / BK;                                 // even for all calls
    async_load_b128(lds0, gbase, W);                       // stage B k-chunk 0
    const bf16_t* ap = A + (size_t)arow * lda + koff;
    v16bf af0 = load_frag_bf16(ap);                        // A k-chunk 0
    v16bf af1;

    for (int i = 0; i < nk; i += 2) {
        gemm_step(i,     nk, &sB[0][0], lds1, gbase, W, ap, l16, koff, af0, af1, acc);
        gemm_step(i + 1, nk, &sB[1][0], lds0, gbase, W, ap, l16, koff, af1, af0, acc);
    }

    // D layout: VGPR r -> row (r + 8*halfsel), col = lane&15 within tile.
#pragma unroll
    for (int r = 0; r < 8; ++r) {
        const int row = mtile + r + halfsel * 8;
#pragma unroll
        for (int t = 0; t < 4; ++t) {
            const int col = nbase + t * 16 + l16;
            if (col < N) {
                float v = acc[t][r];
                if (mode == 1) {
                    v += bias[col];
                    v = (v > 20.0f) ? v : log1pf(__expf(v));  // softplus
                }
                C[(size_t)row * ldc + col] = v;
                if (mode == 2) Cbf[(size_t)row * ldc + col] = (bf16_t)v;
            }
        }
    }
}

// Causal depthwise conv (k=4) + bias + SiLU. Emits f32 (scan input u) and
// bf16 (x-proj GEMM operand) copies.
__global__ __launch_bounds__(256)
void conv_silu_kernel(const float* __restrict__ xr, const float* __restrict__ cw,
                      const float* __restrict__ cb,
                      float* __restrict__ xc_f, bf16_t* __restrict__ xc_bf) {
    const int idx = blockIdx.x * 256 + threadIdx.x;   // over NTOK*D_INNER
    const int d = idx & (D_INNER - 1);
    const int t = idx >> 11;                          // token index b*SEQ+l
    const int l = t & (SEQ - 1);
    const int b = t >> 10;
    float acc = cb[d];
#pragma unroll
    for (int j = 0; j < D_CONV; ++j) {
        const int ll = l - (D_CONV - 1) + j;
        if (ll >= 0)
            acc += xr[((size_t)(b * SEQ + ll)) * (2 * D_INNER) + d] * cw[d * D_CONV + j];
    }
    acc = acc / (1.0f + __expf(-acc));                // SiLU
    xc_f[(size_t)idx] = acc;
    xc_bf[(size_t)idx] = (bf16_t)acc;
}

// Selective scan, state-parallel: 16 lanes per (b,d) channel, lane n owns
// state n; 4-step __shfl_xor reduction of s*C[n] per step. Fuses + u*D and
// the silu(res) gate; writes y directly as bf16 for the out-proj GEMM.
// 65536 threads = 2048 wave32s.
__global__ __launch_bounds__(256)
void selective_scan_kernel(const float* __restrict__ xdbl,   // [NTOK, 96] f32
                           const float* __restrict__ delta,  // [NTOK, D_INNER] f32
                           const float* __restrict__ xc,     // [NTOK, D_INNER] f32
                           const float* __restrict__ xr,     // [NTOK, 2*D_INNER] (res at +D_INNER)
                           const float* __restrict__ A_log,  // [D_INNER, 16]
                           const float* __restrict__ Dp,     // [D_INNER]
                           bf16_t* __restrict__ y) {         // [NTOK, D_INNER] bf16
    const int gid   = blockIdx.x * 256 + threadIdx.x;
    const int n     = gid & (D_STATE - 1);
    const int group = gid >> 4;                // (b,d) channel, 4096 total
    const int b     = group >> 11;
    const int d     = group & (D_INNER - 1);

    const float a  = -__expf(A_log[d * D_STATE + n]);
    const float Dd = Dp[d];
    float s = 0.0f;

    for (int l = 0; l < SEQ; ++l) {
        const int tok = b * SEQ + l;
        const float dv = delta[(size_t)tok * D_INNER + d];       // broadcast in group
        const float u  = xc[(size_t)tok * D_INNER + d];
        const float Bn = xdbl[(size_t)tok * XPROJ_N + DT_RANK + n];
        const float Cn = xdbl[(size_t)tok * XPROJ_N + DT_RANK + D_STATE + n];
        s = __expf(dv * a) * s + dv * Bn * u;
        float sum = s * Cn;
        sum += __shfl_xor(sum, 8, 16);
        sum += __shfl_xor(sum, 4, 16);
        sum += __shfl_xor(sum, 2, 16);
        sum += __shfl_xor(sum, 1, 16);
        if (n == 0) {
            const float rs = xr[(size_t)tok * (2 * D_INNER) + D_INNER + d];
            const float gate = rs / (1.0f + __expf(-rs));        // silu(res)
            y[(size_t)tok * D_INNER + d] = (bf16_t)((sum + u * Dd) * gate);
        }
    }
}

extern "C" void kernel_launch(void* const* d_in, const int* in_sizes, int n_in,
                              void* d_out, int out_size, void* d_ws, size_t ws_size,
                              hipStream_t stream) {
    (void)in_sizes; (void)n_in; (void)out_size; (void)ws_size;
    const float* x       = (const float*)d_in[0];
    const float* in_w    = (const float*)d_in[1];   // [4096, 1024]
    const float* conv_w  = (const float*)d_in[2];   // [2048, 1, 4]
    const float* conv_b  = (const float*)d_in[3];
    const float* xproj_w = (const float*)d_in[4];   // [96, 2048]
    const float* dt_w    = (const float*)d_in[5];   // [2048, 64]
    const float* dt_b    = (const float*)d_in[6];
    const float* A_log   = (const float*)d_in[7];   // [2048, 16]
    const float* Dvec    = (const float*)d_in[8];
    const float* out_w   = (const float*)d_in[9];   // [1024, 2048]
    float* out = (float*)d_out;

    // ---- workspace layout -------------------------------------------------
    float* xr_f    = (float*)d_ws;                          // [NTOK, 4096]
    float* xc_f    = xr_f    + (size_t)NTOK * 2 * D_INNER;  // [NTOK, 2048]
    float* xdbl_f  = xc_f    + (size_t)NTOK * D_INNER;      // [NTOK, 96]
    float* delta_f = xdbl_f  + (size_t)NTOK * XPROJ_N;      // [NTOK, 2048]
    bf16_t* x_bf     = (bf16_t*)(delta_f + (size_t)NTOK * D_INNER);
    bf16_t* inw_bf   = x_bf     + (size_t)NTOK * DIM;           // [4096,1024]
    bf16_t* xproj_bf = inw_bf   + (size_t)2 * D_INNER * DIM;    // [96,2048]
    bf16_t* dtw_bf   = xproj_bf + (size_t)XPROJ_N * D_INNER;    // [2048,64]
    bf16_t* outw_bf  = dtw_bf   + (size_t)D_INNER * DT_RANK;    // [1024,2048]
    bf16_t* xc_bf    = outw_bf  + (size_t)DIM * D_INNER;        // [NTOK,2048]
    bf16_t* xdbl_bf  = xc_bf    + (size_t)NTOK * D_INNER;       // [NTOK,96]
    bf16_t* y_bf     = xdbl_bf  + (size_t)NTOK * XPROJ_N;       // [NTOK,2048]

    const dim3 blk(256);
    auto cvt = [&](const float* s, bf16_t* dst, int n) {
        cvt8_f32_bf16<<<(n / 8 + 255) / 256, blk, 0, stream>>>(s, dst, n);
    };

    // 0) bf16 staging of x + all GEMM weights (one pass, L2-resident after)
    cvt(x,       x_bf,     NTOK * DIM);
    cvt(in_w,    inw_bf,   2 * D_INNER * DIM);
    cvt(xproj_w, xproj_bf, XPROJ_N * D_INNER);
    cvt(dt_w,    dtw_bf,   D_INNER * DT_RANK);
    cvt(out_w,   outw_bf,  DIM * D_INNER);

    // 1) xr = x @ in_w^T            (M=2048, N=4096, K=1024)
    gemm_bf16_wmma<<<dim3(NTOK / 128, (2 * D_INNER) / BN), blk, 0, stream>>>(
        x_bf, inw_bf, xr_f, nullptr, nullptr,
        NTOK, 2 * D_INNER, DIM, DIM, DIM, 2 * D_INNER, 0);

    // 2) causal depthwise conv + SiLU -> xc (f32 + bf16)
    conv_silu_kernel<<<(NTOK * D_INNER) / 256, blk, 0, stream>>>(
        xr_f, conv_w, conv_b, xc_f, xc_bf);

    // 3) x_dbl = xc @ xproj_w^T     (M=2048, N=96, K=2048), dual f32+bf16 out
    gemm_bf16_wmma<<<dim3(NTOK / 128, (XPROJ_N + BN - 1) / BN), blk, 0, stream>>>(
        xc_bf, xproj_bf, xdbl_f, xdbl_bf, nullptr,
        NTOK, XPROJ_N, D_INNER, D_INNER, D_INNER, XPROJ_N, 2);

    // 4) delta = softplus(x_dbl[:, :64] @ dt_w^T + dt_b)  (M=2048, N=2048, K=64)
    gemm_bf16_wmma<<<dim3(NTOK / 128, D_INNER / BN), blk, 0, stream>>>(
        xdbl_bf, dtw_bf, delta_f, nullptr, dt_b,
        NTOK, D_INNER, DT_RANK, XPROJ_N, DT_RANK, D_INNER, 1);

    // 5) selective scan (+ skip, + silu(res) gate) -> y (bf16)
    selective_scan_kernel<<<256, blk, 0, stream>>>(
        xdbl_f, delta_f, xc_f, xr_f, A_log, Dvec, y_bf);

    // 6) out = y @ out_w^T          (M=2048, N=1024, K=2048)
    gemm_bf16_wmma<<<dim3(NTOK / 128, DIM / BN), blk, 0, stream>>>(
        y_bf, outw_bf, out, nullptr, nullptr,
        NTOK, DIM, D_INNER, D_INNER, D_INNER, DIM, 0);
}
